// HyperPatchInvertedResidual_29961691857145
// MI455X (gfx1250) — matrix operand order repaired and checked by reading, not compiled, not run on hardware
//
#include <hip/hip_runtime.h>
#include <hip/hip_bf16.h>

typedef __attribute__((ext_vector_type(16))) _Float16 v16h;
typedef __attribute__((ext_vector_type(8)))  float    v8f;

// Problem constants
#define INC   32
#define OUTC  32
#define HID   192
#define GRP   1024      // 16 batches * 8*8 grid cells
#define KH    18        // patch height (16+2)
#define NPIX1 324       // 18*18
#define NPAD1 336       // padded to multiple of 16
#define NPIX2 256       // 16*16
#define HP    14016     // hyper params per group
#define W2OFF 6144      // INC*HID
#define W3OFF 7872      // + HID*9

__device__ __forceinline__ float relu6f(float v) {
    return fminf(fmaxf(v, 0.0f), 6.0f);
}

// Fragment-ready LDS swizzle (wave32 f16 WMMA layouts, ISA 7.12.2):
//   A 16x32: lane = (row&15) | (bit3(k)<<4),  half j = (k&7) | ((k&16)>>1)
//   B 32x16: lane = (col&15) | (bit3(k)<<4),  half j = (k&7) | ((k&16)>>1)
// so each lane's 16 halfs are one contiguous v16h -> 2x ds_load_b128.

// ---------------------------------------------------------------------------
// Kernel 1: per-group GEMM1  y = w1 (192x32) @ patch (32x324), one
// v_wmma_f32_16x16x32_f16 per 16x16 tile (K=32 exactly). 384 thr = 12 waves,
// one M-tile per wave; BN1 raw stats held in registers across all 21 N-tiles.
// ---------------------------------------------------------------------------
__global__ __launch_bounds__(384) void
k1_gemm1(const float* __restrict__ x, const float* __restrict__ s,
         float* __restrict__ yb, float* __restrict__ sum1, float* __restrict__ sq1) {
    const int g = blockIdx.x;
    const int b = g >> 6;
    const int f = g & 63;
    const int fi = f >> 3;
    const int fj = f & 7;
    const int tid = threadIdx.x;

    __shared__ v16h wAf[12 * 32];   // 12 KB  A fragments [mt][lane]
    __shared__ v16h wBf[21 * 32];   // 21 KB  B fragments [nt][lane]
    _Float16* wAh = (_Float16*)wAf;
    _Float16* wBh = (_Float16*)wBf;

    const size_t sbase = (size_t)b * HP * 64 + f;
    // stage w1 -> fragment-ready LDS (w1 flat k == o*32+c == idx)
    for (int idx = tid; idx < HID * INC; idx += 384) {
        int o = idx >> 5, c = idx & 31;
        int mt = o >> 4;
        int lane = (o & 15) | (((c >> 3) & 1) << 4);
        int j = (c & 7) | ((c & 16) >> 1);
        wAh[(mt * 32 + lane) * 16 + j] = (_Float16)s[sbase + (size_t)idx * 64];
    }
    // stage 18x18 reflect-padded patch -> fragment-ready LDS (pad cols = 0)
    const size_t xbase = (size_t)b * INC * 128 * 128;
    for (int idx = tid; idx < INC * NPAD1; idx += 384) {
        int c = idx / NPAD1, pix = idx - c * NPAD1;
        float val = 0.0f;
        if (pix < NPIX1) {
            int i = pix / KH, jj = pix - i * KH;
            int r = fi * 16 + i - 1;   r = (r < 0) ? -r : ((r > 127) ? 254 - r : r);
            int q = fj * 16 + jj - 1;  q = (q < 0) ? -q : ((q > 127) ? 254 - q : q);
            val = x[xbase + ((size_t)c * 128 + r) * 128 + q];
        }
        int nt = pix >> 4;
        int lane = (pix & 15) | (((c >> 3) & 1) << 4);
        int j = (c & 7) | ((c & 16) >> 1);
        wBh[(nt * 32 + lane) * 16 + j] = (_Float16)val;
    }
    __syncthreads();

    const int wave = tid >> 5, lane = tid & 31;
    const int row16 = lane & 15, hi = lane >> 4;
    const int mt = wave;                       // 12 waves == 12 M-tiles

    float ss[8], qq[8];
#pragma unroll
    for (int v = 0; v < 8; ++v) { ss[v] = 0.0f; qq[v] = 0.0f; }

    const v16h a = wAf[mt * 32 + lane];        // 2x ds_load_b128, reused 21x
    for (int nt = 0; nt < 21; ++nt) {
        const v16h bf = wBf[nt * 32 + lane];
        v8f acc = {};
        acc = __builtin_amdgcn_wmma_f32_16x16x32_f16(false, a, false, bf,
                                                     (short)0, acc, false, false);
#pragma unroll
        for (int v = 0; v < 8; ++v) {
            const int ch = mt * 16 + (hi << 3) + v;
            const float val = acc[v];
            yb[((size_t)g * HID + ch) * NPAD1 + nt * 16 + row16] = val;
            ss[v] += val;
            qq[v] += val * val;                // pad cols multiply zero B -> add 0
        }
    }
    // one 16-lane tree reduction per wave (not per tile)
#pragma unroll
    for (int v = 0; v < 8; ++v) {
        float s1 = ss[v], s2 = qq[v];
#pragma unroll
        for (int m = 1; m < 16; m <<= 1) {
            s1 += __shfl_xor(s1, m, 32);
            s2 += __shfl_xor(s2, m, 32);
        }
        if (row16 == 0) {
            const int ch = mt * 16 + (hi << 3) + v;
            atomicAdd(&sum1[ch], s1);
            atomicAdd(&sq1[ch], s2);
        }
    }
}

// ---------------------------------------------------------------------------
// BN stats finalize: scale = gamma * rsqrt(var+eps), shift = beta - mean*scale
// ---------------------------------------------------------------------------
__global__ void stats_finalize(const float* __restrict__ sum, const float* __restrict__ sq,
                               const float* __restrict__ gamma, const float* __restrict__ beta,
                               float* __restrict__ scale, float* __restrict__ shift,
                               int nch, float invN) {
    int i = threadIdx.x;
    if (i < nch) {
        float mean = sum[i] * invN;
        float var  = sq[i] * invN - mean * mean;
        float sc   = gamma[i] * rsqrtf(var + 1e-5f);
        scale[i] = sc;
        shift[i] = beta[i] - mean * sc;
    }
}

// ---------------------------------------------------------------------------
// Kernel 3: BN1+ReLU6 fused into depthwise 3x3 valid conv; BN2 raw stats.
// ---------------------------------------------------------------------------
__global__ __launch_bounds__(256) void
k3_depthwise(const float* __restrict__ s, const float* __restrict__ yb,
             const float* __restrict__ scale1, const float* __restrict__ shift1,
             float* __restrict__ y2b, float* __restrict__ sum2, float* __restrict__ sq2) {
    const int g = blockIdx.x;
    const int b = g >> 6;
    const int f = g & 63;
    const int tid = threadIdx.x;

    __shared__ float act[NPIX1];
    __shared__ float taps[9];
    __shared__ float lsum[HID], lsq[HID];
    if (tid < HID) { lsum[tid] = 0.0f; lsq[tid] = 0.0f; }

    const size_t sbase = (size_t)b * HP * 64 + f;
    const int i = tid >> 4, j = tid & 15;

    for (int ch = 0; ch < HID; ++ch) {
        __syncthreads();
        if (tid < 9)
            taps[tid] = s[sbase + (size_t)(W2OFF + ch * 9 + tid) * 64];
        const float sc = scale1[ch], sh = shift1[ch];
        for (int idx = tid; idx < NPIX1; idx += 256) {
            float v = yb[((size_t)g * HID + ch) * NPAD1 + idx];
            act[idx] = relu6f(sc * v + sh);
        }
        __syncthreads();
        float accv = 0.0f;
#pragma unroll
        for (int di = 0; di < 3; ++di)
#pragma unroll
            for (int dj = 0; dj < 3; ++dj)
                accv += act[(i + di) * KH + (j + dj)] * taps[di * 3 + dj];
        y2b[((size_t)g * HID + ch) * NPIX2 + tid] = accv;
        float s1 = accv, s2 = accv * accv;
#pragma unroll
        for (int m = 1; m < 32; m <<= 1) {
            s1 += __shfl_xor(s1, m, 32);
            s2 += __shfl_xor(s2, m, 32);
        }
        if ((tid & 31) == 0) { atomicAdd(&lsum[ch], s1); atomicAdd(&lsq[ch], s2); }
    }
    __syncthreads();
    if (tid < HID) { atomicAdd(&sum2[tid], lsum[tid]); atomicAdd(&sq2[tid], lsq[tid]); }
}

// ---------------------------------------------------------------------------
// Kernel 5: BN2+ReLU6 fused into f16 convert; GEMM3 y3 = w3 (32x192) @ y2
// (192x256), K chunked by 32 through fragment-ready LDS; per-wave VGPR
// accumulators across the whole K loop; BN3 stats reduced once at the end.
// ---------------------------------------------------------------------------
__global__ __launch_bounds__(256) void
k5_gemm3(const float* __restrict__ s, const float* __restrict__ y2b,
         const float* __restrict__ scale2, const float* __restrict__ shift2,
         float* __restrict__ y3b, float* __restrict__ sum3, float* __restrict__ sq3) {
    const int g = blockIdx.x;
    const int b = g >> 6;
    const int f = g & 63;
    const int tid = threadIdx.x;

    __shared__ v16h wA3f[2 * 32 * 6];   // 12 KB  A fragments [mt][lane][kc]
    __shared__ v16h Bcf[16 * 32];       // 16 KB  B chunk fragments [nt][lane]
    __shared__ float lsum[OUTC], lsq[OUTC];
    _Float16* wA3h = (_Float16*)wA3f;
    _Float16* Bch  = (_Float16*)Bcf;

    const size_t sbase = (size_t)b * HP * 64 + f;
    for (int idx = tid; idx < OUTC * HID; idx += 256) {
        int o = idx / HID, k = idx - o * HID;       // w3 flat = o*192 + k
        int mt = o >> 4;
        int lane = (o & 15) | (((k >> 3) & 1) << 4);
        int kc = k >> 5, kk = k & 31;
        int j = (kk & 7) | ((kk & 16) >> 1);
        wA3h[((mt * 32 + lane) * 6 + kc) * 16 + j] =
            (_Float16)s[sbase + (size_t)(W3OFF + idx) * 64];
    }
    if (tid < OUTC) { lsum[tid] = 0.0f; lsq[tid] = 0.0f; }

    const int wave = tid >> 5, lane = tid & 31;
    const int row16 = lane & 15, hi = lane >> 4;

    v8f acc[4] = {};
    for (int kc = 0; kc < 6; ++kc) {
        __syncthreads();
        for (int idx = tid; idx < 32 * NPIX2; idx += 256) {
            int c = idx >> 8, pix = idx & 255;
            int ch = kc * 32 + c;
            float v = y2b[((size_t)g * HID + ch) * NPIX2 + pix];
            int bl = (pix & 15) | (((c >> 3) & 1) << 4);
            Bch[(((pix >> 4) * 32 + bl) * 16) + ((c & 7) | ((c & 16) >> 1))] =
                (_Float16)relu6f(scale2[ch] * v + shift2[ch]);
        }
        __syncthreads();
#pragma unroll
        for (int tt = 0; tt < 4; ++tt) {
            const int id = wave + tt * 8;            // 32 tiles: mt in {0,1}, nt 0..15
            const int mt = id >> 4, nt = id & 15;
            const v16h a  = wA3f[(mt * 32 + lane) * 6 + kc];
            const v16h bf = Bcf[nt * 32 + lane];
            acc[tt] = __builtin_amdgcn_wmma_f32_16x16x32_f16(false, a, false, bf,
                                                             (short)0, acc[tt], false, false);
        }
    }
    __syncthreads();
#pragma unroll
    for (int tt = 0; tt < 4; ++tt) {
        const int id = wave + tt * 8;
        const int mt = id >> 4, nt = id & 15;
#pragma unroll
        for (int v = 0; v < 8; ++v) {
            const int ch = mt * 16 + (hi << 3) + v;
            const float val = acc[tt][v];
            y3b[((size_t)g * OUTC + ch) * NPIX2 + nt * 16 + row16] = val;
            float s1 = val, s2 = val * val;
#pragma unroll
            for (int m = 1; m < 16; m <<= 1) {
                s1 += __shfl_xor(s1, m, 32);
                s2 += __shfl_xor(s2, m, 32);
            }
            if (row16 == 0) { atomicAdd(&lsum[ch], s1); atomicAdd(&lsq[ch], s2); }
        }
    }
    __syncthreads();
    if (tid < OUTC) { atomicAdd(&sum3[tid], lsum[tid]); atomicAdd(&sq3[tid], lsq[tid]); }
}

// ---------------------------------------------------------------------------
// Kernel 7: BN3 + spatial reassembly + residual add.
// ---------------------------------------------------------------------------
__global__ __launch_bounds__(256) void
k7_final(const float* __restrict__ x, const float* __restrict__ y3b,
         const float* __restrict__ scale3, const float* __restrict__ shift3,
         float* __restrict__ out) {
    const size_t t = (size_t)blockIdx.x * 256 + threadIdx.x;   // exactly 16*32*128*128
    const int w  = (int)(t & 127);
    const int h  = (int)((t >> 7) & 127);
    const int oc = (int)((t >> 14) & 31);
    const int b  = (int)(t >> 19);
    const int g   = b * 64 + (h >> 4) * 8 + (w >> 4);
    const int pix = (h & 15) * 16 + (w & 15);
    const float v = y3b[((size_t)g * OUTC + oc) * NPIX2 + pix];
    out[t] = x[t] + scale3[oc] * v + shift3[oc];
}

// ---------------------------------------------------------------------------
extern "C" void kernel_launch(void* const* d_in, const int* in_sizes, int n_in,
                              void* d_out, int out_size, void* d_ws, size_t ws_size,
                              hipStream_t stream) {
    const float* x  = (const float*)d_in[0];
    const float* s  = (const float*)d_in[1];
    const float* g1 = (const float*)d_in[2];
    const float* b1 = (const float*)d_in[3];
    const float* g2 = (const float*)d_in[4];
    const float* b2 = (const float*)d_in[5];
    const float* g3 = (const float*)d_in[6];
    const float* b3 = (const float*)d_in[7];
    float* out = (float*)d_out;

    float* ws  = (float*)d_ws;
    float* yb  = ws;                                       // 1024*192*336 fp32 (~264 MB)
    float* y2b = yb  + (size_t)GRP * HID * NPAD1;          // 1024*192*256 fp32 (~201 MB)
    float* y3b = y2b + (size_t)GRP * HID * NPIX2;          // 1024*32*256 fp32 (~33.5 MB)
    float* st  = y3b + (size_t)GRP * OUTC * NPIX2;         // stats: 1664 floats
    float* sum1 = st,        *sq1 = st + 192, *scale1 = st + 384,  *shift1 = st + 576;
    float* sum2 = st + 768,  *sq2 = st + 960, *scale2 = st + 1152, *shift2 = st + 1344;
    float* sum3 = st + 1536, *sq3 = st + 1568,*scale3 = st + 1600, *shift3 = st + 1632;

    hipMemsetAsync(st, 0, 1664 * sizeof(float), stream);

    k1_gemm1<<<GRP, 384, 0, stream>>>(x, s, yb, sum1, sq1);
    stats_finalize<<<1, 256, 0, stream>>>(sum1, sq1, g1, b1, scale1, shift1, HID,
                                          1.0f / (float)(GRP * NPIX1));
    k3_depthwise<<<GRP, 256, 0, stream>>>(s, yb, scale1, shift1, y2b, sum2, sq2);
    stats_finalize<<<1, 256, 0, stream>>>(sum2, sq2, g2, b2, scale2, shift2, HID,
                                          1.0f / (float)(GRP * NPIX2));
    k5_gemm3<<<GRP, 256, 0, stream>>>(s, y2b, scale2, shift2, y3b, sum3, sq3);
    stats_finalize<<<1, 256, 0, stream>>>(sum3, sq3, g3, b3, scale3, shift3, OUTC,
                                          1.0f / (float)(GRP * NPIX2));
    k7_final<<<32768, 256, 0, stream>>>(x, y3b, scale3, shift3, out);
}
